// PrunedAttention_36636071035070
// MI455X (gfx1250) — matrix operand order repaired and checked by reading, not compiled, not run on hardware
//
#include <hip/hip_runtime.h>

// ---------------------------------------------------------------------------
// Pruned attention (B=4, N=2304, C=768, nh=8, hd=48) for gfx1250 (MI455X).
// bf16 WMMA (v_wmma_f32_16x16x32_bf16) everywhere, f32 accumulation.
// ---------------------------------------------------------------------------

typedef __bf16 bf16_t;
typedef __attribute__((ext_vector_type(16))) __bf16 v16bf;
typedef __attribute__((ext_vector_type(8)))  __bf16 v8bf;
typedef __attribute__((ext_vector_type(8)))  float  v8f;

#define WMMA_BF16(a, b, c) \
  __builtin_amdgcn_wmma_f32_16x16x32_bf16(false, (a), false, (b), (short)0, (c), false, false)

// Problem constants
#define BATCH 4
#define NH    8
#define SEQ   2304      // 48*48
#define CDIM  768
#define AHD   384       // nh*hd
#define HD    48
#define HDP   64        // padded head dim for QK^T
#define MROWS 9216      // BATCH*SEQ
#define QSCALE 0.14433756729740643f  // 1/sqrt(48)

// A-operand tile (16x32, bf16): lane = row m (0..15), K chunks {kb..kb+7, kb+16..kb+23}
static __device__ __forceinline__ v16bf load_A16x32(const bf16_t* __restrict__ p, int lane, int ld) {
  const int m  = lane & 15;
  const int kb = (lane >> 4) << 3;
  v8bf lo = *(const v8bf*)(p + (size_t)m * ld + kb);
  v8bf hi = *(const v8bf*)(p + (size_t)m * ld + kb + 16);
  return __builtin_shufflevector(lo, hi, 0,1,2,3,4,5,6,7,8,9,10,11,12,13,14,15);
}

// B-operand tile (32x16, bf16) read from B^T-major memory (row n = output col,
// contiguous K): lane = col n, K chunks {kb..kb+7, kb+16..kb+23}, kb = 8*(lane>=16).
static __device__ __forceinline__ v16bf load_B32x16(const bf16_t* __restrict__ pT, int lane, int ld) {
  const int n  = lane & 15;
  const int kb = (lane >> 4) << 3;
  v8bf lo = *(const v8bf*)(pT + (size_t)n * ld + kb);
  v8bf hi = *(const v8bf*)(pT + (size_t)n * ld + kb + 16);
  return __builtin_shufflevector(lo, hi, 0,1,2,3,4,5,6,7,8,9,10,11,12,13,14,15);
}

// ---------------------------------------------------------------------------
// Kernel 1: f32 -> bf16 convert
// ---------------------------------------------------------------------------
__global__ void cvt_f32_bf16_kernel(const float* __restrict__ src, bf16_t* __restrict__ dst, int n) {
  int i = blockIdx.x * blockDim.x + threadIdx.x;
  if (i < n) dst[i] = (bf16_t)src[i];
}

// ---------------------------------------------------------------------------
// Kernel 2: fused QKV projection. One wave -> 16 rows x 64 cols of one of q/k/v.
//   q_raw = x*Wq^T + qb (f32), k_raw = x*Wk^T (f32),
//   vT[b][h][d][n] = (x*Wv^T + vb) (bf16, transposed for PV B-operand)
// grid = (576 rowblocks, 18 tasks: mat*6 + colblock64), block = 32
// ---------------------------------------------------------------------------
__global__ void qkv_gemm_kernel(const bf16_t* __restrict__ xb,
                                const bf16_t* __restrict__ wqb,
                                const bf16_t* __restrict__ wkb,
                                const bf16_t* __restrict__ wvb,
                                const float* __restrict__ qbias,
                                const float* __restrict__ vbias,
                                float* __restrict__ qraw,
                                float* __restrict__ kraw,
                                bf16_t* __restrict__ vT) {
  const int lane    = threadIdx.x;
  const int r       = blockIdx.x;        // 16-row block of flattened [9216, 768]
  const int task    = blockIdx.y;        // 0..17
  const int mat     = task / 6;          // 0=q 1=k 2=v
  const int colbase = (task % 6) * 64;
  const bf16_t* wb = (mat == 0) ? wqb : (mat == 1) ? wkb : wvb;
  const bf16_t* arow = xb + (size_t)r * 16 * CDIM;

  v8f acc[4] = {v8f{}, v8f{}, v8f{}, v8f{}};
  for (int k = 0; k < CDIM; k += 32) {
    v16bf a = load_A16x32(arow + k, lane, CDIM);
#pragma unroll
    for (int t = 0; t < 4; ++t) {
      v16bf bt = load_B32x16(wb + (size_t)(colbase + t * 16) * CDIM + k, lane, CDIM);
      acc[t] = WMMA_BF16(a, bt, acc[t]);
    }
  }

  const int mrow = (lane >> 4) * 8;
  const int ncol = lane & 15;
#pragma unroll
  for (int t = 0; t < 4; ++t) {
    const int col = colbase + t * 16 + ncol;
#pragma unroll
    for (int j = 0; j < 8; ++j) {
      const int grow = r * 16 + mrow + j;
      const float val = acc[t][j];
      if (mat == 0) {
        qraw[(size_t)grow * AHD + col] = val + qbias[col];
      } else if (mat == 1) {
        kraw[(size_t)grow * AHD + col] = val;
      } else {
        const float vv = val + vbias[col];
        const int b = grow / SEQ, n = grow % SEQ;
        const int h = col / HD,  d = col % HD;
        vT[((size_t)(b * NH + h) * HD + d) * SEQ + n] = (bf16_t)vv;
      }
    }
  }
}

// ---------------------------------------------------------------------------
// Kernel 3: RoPE (rotate-half) + q-scale + pad hd 48->64.
//   q_ws/k_ws: bf16 [B][nh][N][64], q pre-scaled by 1/sqrt(hd).
// One thread per (b,h,n,i), i in 0..31: i<24 -> rotation pair (i, i+24),
// i>=24 -> zero-fill pad elements 48+2*(i-24), +1.
// ---------------------------------------------------------------------------
__global__ void rope_kernel(const float* __restrict__ rope_cos,
                            const float* __restrict__ rope_sin,
                            const float* __restrict__ qraw,
                            const float* __restrict__ kraw,
                            bf16_t* __restrict__ qws,
                            bf16_t* __restrict__ kws) {
  const int tid = blockIdx.x * blockDim.x + threadIdx.x;  // < 4*8*2304*32
  const int i  = tid & 31;
  int rest     = tid >> 5;
  const int n  = rest % SEQ;
  rest /= SEQ;
  const int h  = rest & 7;
  const int b  = rest >> 3;

  const size_t g    = (size_t)b * SEQ + n;
  const size_t src  = g * AHD + h * HD;
  const size_t dst  = ((size_t)(b * NH + h) * SEQ + n) * HDP;

  if (i < 24) {
    const int d1 = i, d2 = i + 24;
    const float c1 = rope_cos[(size_t)n * HD + d1];
    const float s1 = rope_sin[(size_t)n * HD + d1];
    const float c2 = rope_cos[(size_t)n * HD + d2];
    const float s2 = rope_sin[(size_t)n * HD + d2];
    const float q1 = qraw[src + d1], q2 = qraw[src + d2];
    const float k1 = kraw[src + d1], k2 = kraw[src + d2];
    qws[dst + d1] = (bf16_t)((q1 * c1 - q2 * s1) * QSCALE);
    qws[dst + d2] = (bf16_t)((q2 * c2 + q1 * s2) * QSCALE);
    kws[dst + d1] = (bf16_t)(k1 * c1 - k2 * s1);
    kws[dst + d2] = (bf16_t)(k2 * c2 + k1 * s2);
  } else {
    const int d = HD + (i - 24) * 2;
    qws[dst + d] = (bf16_t)0.0f; qws[dst + d + 1] = (bf16_t)0.0f;
    kws[dst + d] = (bf16_t)0.0f; kws[dst + d + 1] = (bf16_t)0.0f;
  }
}

// ---------------------------------------------------------------------------
// Kernel 4: flash attention. One wave = 16 query rows of one (b,h).
// Key loop in 32-key blocks: S = Q*K^T via 4 WMMA, online softmax with
// 16-lane shfl reductions (row r = j + 8*(lane>=16) matches accumulator slots),
// P staged through LDS (C-layout -> A-layout), PV via 3 WMMA (hd=48).
// grid = 576, block = 256 (8 independent waves, disjoint LDS slices).
// ---------------------------------------------------------------------------
__global__ __launch_bounds__(256) void attn_kernel(const bf16_t* __restrict__ qws,
                                                   const bf16_t* __restrict__ kws,
                                                   const bf16_t* __restrict__ vT,
                                                   bf16_t* __restrict__ attnout) {
  __shared__ bf16_t Plds[8 * 16 * 32];   // per-wave 16x32 bf16 P staging
  const int lane = threadIdx.x & 31;
  const int wid  = threadIdx.x >> 5;
  const int task = blockIdx.x * 8 + wid;             // < 4608
  const int rb   = task % 144;
  const int h    = (task / 144) & 7;
  const int b    = task / 1152;
  const size_t bh = (size_t)b * NH + h;

  const bf16_t* qbase = qws + (bh * SEQ + (size_t)rb * 16) * HDP;
  const bf16_t* kbase = kws + bh * SEQ * HDP;
  const bf16_t* vbase = vT + bh * HD * SEQ;

  const v16bf qa0 = load_A16x32(qbase, lane, HDP);        // K = 0..31 of head dim
  const v16bf qa1 = load_A16x32(qbase + 32, lane, HDP);   // K = 32..63 (zero padded)

  float mrun[8], lrun[8];
  v8f accO[3] = {v8f{}, v8f{}, v8f{}};
#pragma unroll
  for (int j = 0; j < 8; ++j) { mrun[j] = -1.0e30f; lrun[j] = 0.0f; }

  const int pbase = wid * 512;
  const int m8    = lane & 15;           // A-layout row for P reload
  const int kb8   = (lane >> 4) << 3;    // A-layout K chunk base
  const int col   = lane & 15;           // C-layout column
  const int rbase = (lane >> 4) * 8;     // C-layout row base

  for (int kb = 0; kb < SEQ; kb += 32) {
    // next K-block prefetch (speculative; harmless at tail)
    __builtin_prefetch(kbase + (size_t)(kb + 32) * HDP, 0, 1);

    // ---- S = (q*scale) . K^T : two 16-key output tiles, K-dim 64 ----
    v8f s0 = {}, s1 = {};
    {
      v16bf k00 = load_B32x16(kbase + (size_t)(kb +  0) * HDP +  0, lane, HDP);
      v16bf k01 = load_B32x16(kbase + (size_t)(kb +  0) * HDP + 32, lane, HDP);
      v16bf k10 = load_B32x16(kbase + (size_t)(kb + 16) * HDP +  0, lane, HDP);
      v16bf k11 = load_B32x16(kbase + (size_t)(kb + 16) * HDP + 32, lane, HDP);
      s0 = WMMA_BF16(qa0, k00, s0);
      s0 = WMMA_BF16(qa1, k01, s0);
      s1 = WMMA_BF16(qa0, k10, s1);
      s1 = WMMA_BF16(qa1, k11, s1);
    }

    // ---- online softmax update ----
    float p0[8], p1[8];
#pragma unroll
    for (int j = 0; j < 8; ++j) {
      float v = fmaxf(s0[j], s1[j]);
      v = fmaxf(v, __shfl_xor(v, 1, 32));
      v = fmaxf(v, __shfl_xor(v, 2, 32));
      v = fmaxf(v, __shfl_xor(v, 4, 32));
      v = fmaxf(v, __shfl_xor(v, 8, 32));           // block row max
      const float nm   = fmaxf(mrun[j], v);
      const float corr = __expf(mrun[j] - nm);
      mrun[j] = nm;
      p0[j] = __expf(s0[j] - nm);
      p1[j] = __expf(s1[j] - nm);
      float rs = p0[j] + p1[j];
      rs += __shfl_xor(rs, 1, 32);
      rs += __shfl_xor(rs, 2, 32);
      rs += __shfl_xor(rs, 4, 32);
      rs += __shfl_xor(rs, 8, 32);                  // block row sum
      lrun[j] = lrun[j] * corr + rs;
      accO[0][j] *= corr; accO[1][j] *= corr; accO[2][j] *= corr;
    }

    // ---- stage P (C-layout) into LDS as row-major 16x32 bf16 ----
#pragma unroll
    for (int j = 0; j < 8; ++j) {
      Plds[pbase + (rbase + j) * 32 + col]      = (bf16_t)p0[j];
      Plds[pbase + (rbase + j) * 32 + col + 16] = (bf16_t)p1[j];
    }
    asm volatile("s_wait_dscnt 0" ::: "memory");   // cross-lane LDS RAW within wave

    // ---- reload P as A-operand (16x32) ----
    v8bf plo = *(const v8bf*)&Plds[pbase + m8 * 32 + kb8];
    v8bf phi = *(const v8bf*)&Plds[pbase + m8 * 32 + kb8 + 16];
    v16bf pa = __builtin_shufflevector(plo, phi, 0,1,2,3,4,5,6,7,8,9,10,11,12,13,14,15);

    // ---- O += P * V  (B^T rows = vT[d][n]) ----
#pragma unroll
    for (int t = 0; t < 3; ++t) {
      v16bf vb = load_B32x16(vbase + (size_t)(t * 16) * SEQ + kb, lane, SEQ);
      accO[t] = WMMA_BF16(pa, vb, accO[t]);
    }
  }

  // ---- finalize: divide by row sums, write bf16 [B, N, 384] ----
#pragma unroll
  for (int t = 0; t < 3; ++t) {
    const int c = h * HD + t * 16 + col;
#pragma unroll
    for (int j = 0; j < 8; ++j) {
      const int n = rb * 16 + rbase + j;
      const float o = accO[t][j] / lrun[j];
      attnout[((size_t)b * SEQ + n) * AHD + c] = (bf16_t)o;
    }
  }
}

// ---------------------------------------------------------------------------
// Kernel 5: output projection: out = attnout * Wp^T + bp  (f32 result)
// grid = (576 rowblocks, 12 colblocks64), block = 32
// ---------------------------------------------------------------------------
__global__ void outproj_kernel(const bf16_t* __restrict__ attnout,
                               const bf16_t* __restrict__ wpb,
                               const float* __restrict__ bp,
                               float* __restrict__ out) {
  const int lane = threadIdx.x;
  const int r  = blockIdx.x;
  const int cb = blockIdx.y * 64;
  const bf16_t* arow = attnout + (size_t)r * 16 * AHD;

  v8f acc[4] = {v8f{}, v8f{}, v8f{}, v8f{}};
  for (int k = 0; k < AHD; k += 32) {
    v16bf a = load_A16x32(arow + k, lane, AHD);
#pragma unroll
    for (int t = 0; t < 4; ++t) {
      v16bf bt = load_B32x16(wpb + (size_t)(cb + t * 16) * AHD + k, lane, AHD);
      acc[t] = WMMA_BF16(a, bt, acc[t]);
    }
  }

  const int mrow = (lane >> 4) * 8;
  const int ncol = lane & 15;
#pragma unroll
  for (int t = 0; t < 4; ++t) {
    const int c = cb + t * 16 + ncol;
    const float bias = bp[c];
#pragma unroll
    for (int j = 0; j < 8; ++j) {
      const int g = r * 16 + mrow + j;
      out[(size_t)g * CDIM + c] = acc[t][j] + bias;
    }
  }
}

// ---------------------------------------------------------------------------
// Host launch
// ---------------------------------------------------------------------------
extern "C" void kernel_launch(void* const* d_in, const int* in_sizes, int n_in,
                              void* d_out, int out_size, void* d_ws, size_t ws_size,
                              hipStream_t stream) {
  const float* x        = (const float*)d_in[0];
  const float* rope_cos = (const float*)d_in[1];
  const float* rope_sin = (const float*)d_in[2];
  const float* Wq       = (const float*)d_in[3];
  const float* Wk       = (const float*)d_in[4];
  const float* Wv       = (const float*)d_in[5];
  const float* q_bias   = (const float*)d_in[6];
  const float* v_bias   = (const float*)d_in[7];
  const float* Wp       = (const float*)d_in[8];
  const float* bp       = (const float*)d_in[9];
  float* out = (float*)d_out;
  char* ws = (char*)d_ws;

  // workspace layout (bytes, all offsets 256-aligned)
  const size_t NX = (size_t)MROWS * CDIM;   // 7,077,888
  const size_t NW = (size_t)AHD * CDIM;     //   294,912
  size_t off = 0;
  bf16_t* xb   = (bf16_t*)(ws + off); off += NX * 2;            // x bf16
  bf16_t* wqb  = (bf16_t*)(ws + off); off += NW * 2;
  bf16_t* wkb  = (bf16_t*)(ws + off); off += NW * 2;
  bf16_t* wvb  = (bf16_t*)(ws + off); off += NW * 2;
  bf16_t* wpb  = (bf16_t*)(ws + off); off += NW * 2;
  float*  qraw = (float*)(ws + off);  size_t qraw_off = off; off += (size_t)MROWS * AHD * 4;
  float*  kraw = (float*)(ws + off);  off += (size_t)MROWS * AHD * 4;
  bf16_t* qws  = (bf16_t*)(ws + off); off += (size_t)BATCH * NH * SEQ * HDP * 2;
  bf16_t* kws  = (bf16_t*)(ws + off); off += (size_t)BATCH * NH * SEQ * HDP * 2;
  bf16_t* vT   = (bf16_t*)(ws + off); off += (size_t)BATCH * NH * HD * SEQ * 2;
  // attn output reuses qraw (dead after rope_kernel); needs 9216*384*2 of its 9216*384*4 bytes
  bf16_t* attnb = (bf16_t*)(ws + qraw_off);
  (void)ws_size; (void)in_sizes; (void)n_in; (void)out_size;

  // 1) converts
  cvt_f32_bf16_kernel<<<(int)((NX + 255) / 256), 256, 0, stream>>>(x, xb, (int)NX);
  cvt_f32_bf16_kernel<<<(int)((NW + 255) / 256), 256, 0, stream>>>(Wq, wqb, (int)NW);
  cvt_f32_bf16_kernel<<<(int)((NW + 255) / 256), 256, 0, stream>>>(Wk, wkb, (int)NW);
  cvt_f32_bf16_kernel<<<(int)((NW + 255) / 256), 256, 0, stream>>>(Wv, wvb, (int)NW);
  cvt_f32_bf16_kernel<<<(int)((NW + 255) / 256), 256, 0, stream>>>(Wp, wpb, (int)NW);

  // 2) QKV projections
  qkv_gemm_kernel<<<dim3(MROWS / 16, 18), 32, 0, stream>>>(
      xb, wqb, wkb, wvb, q_bias, v_bias, qraw, kraw, vT);

  // 3) RoPE + scale + pad
  rope_kernel<<<(BATCH * NH * SEQ * 32) / 256, 256, 0, stream>>>(
      rope_cos, rope_sin, qraw, kraw, qws, kws);

  // 4) flash attention
  attn_kernel<<<(BATCH * NH * (SEQ / 16)) / 8, 256, 0, stream>>>(qws, kws, vT, attnb);

  // 5) output projection
  outproj_kernel<<<dim3(MROWS / 16, CDIM / 64), 32, 0, stream>>>(attnb, wpb, bp, out);
}
